// SCFA_66571993088914
// MI455X (gfx1250) — matrix operand order repaired and checked by reading, compiled, not run on hardware
//
#include <hip/hip_runtime.h>
#include <hip/hip_bf16.h>
#include <math.h>

typedef __attribute__((ext_vector_type(16))) _Float16 v16h;
typedef __attribute__((ext_vector_type(8)))  _Float16 v8h;
typedef __attribute__((ext_vector_type(8)))  float    v8f;

#define NSTREAM 2
#define BATCH   8
#define CDIM    256
#define NPIX    4096
#define NHEAD   4
#define HDIM    64
#define O3      768

union HFrag { v16h v; _Float16 h[16]; };
union HPack { v8h v; _Float16 h[8]; };

// A operand, 16x32 f16, source row-major [m][k], row pointer given.
// lane l: m = l&15, g = l>>4. elems 0..7 = k0+8g+0..7 ; elems 8..15 = k0+16+8g+0..7
__device__ inline v16h load_a16(const _Float16* __restrict__ row, int k0, int g) {
    v16h a;
    *(v8h*)&a         = *(const v8h*)(row + k0 + 8 * g);
    *(((v8h*)&a) + 1) = *(const v8h*)(row + k0 + 16 + 8 * g);
    return a;
}

// B operand, 32x16 f16, source transposed-row-major [n][k], row pointer given.
// lane l: n = l&15, g = l>>4. elems 0..7 = k0+8g+0..7 ; elems 8..15 = k0+16+8g+0..7
__device__ inline v16h load_b16(const _Float16* __restrict__ row, int k0, int g) {
    v16h b;
    *(v8h*)&b         = *(const v8h*)(row + k0 + 8 * g);
    *(((v8h*)&b) + 1) = *(const v8h*)(row + k0 + 16 + 8 * g);
    return b;
}

__device__ inline v8f wmma16(v16h a, v16h b, v8f c) {
    return __builtin_amdgcn_wmma_f32_16x16x32_f16(false, a, false, b, (short)0, c, false, false);
}

// ---------------- fp32 -> f16 bulk convert (weight prep; n multiple of 1024)
__global__ __launch_bounds__(256) void cvt16_kernel(const float* __restrict__ src,
                                                    _Float16* __restrict__ dst, int n) {
    int i = (blockIdx.x * 256 + threadIdx.x) * 4;
#pragma unroll
    for (int j = 0; j < 4; ++j) dst[i + j] = (_Float16)src[i + j];
}

// ---------------- BN coefficient prep: alpha = g/sqrt(v+eps), beta = qb*alpha + bb - m*alpha
__global__ __launch_bounds__(768) void bnprep_kernel(
    const float* qb1, const float* g1, const float* bb1, const float* m1, const float* v1,
    const float* qb2, const float* g2, const float* bb2, const float* m2, const float* v2,
    float* alpha, float* beta) {
    int o = threadIdx.x, s = blockIdx.x;
    const float* qb = s ? qb2 : qb1;
    const float* g  = s ? g2  : g1;
    const float* bb = s ? bb2 : bb1;
    const float* m  = s ? m2  : m1;
    const float* v  = s ? v2  : v1;
    float inv = g[o] * rsqrtf(v[o] + 1e-5f);
    alpha[s * O3 + o] = inv;
    beta [s * O3 + o] = qb[o] * inv + bb[o] - m[o] * inv;
}

// ---------------- Multi-scale depthwise conv (fused effective 7x7) -> xfT [s][b][n][c] f16
__global__ __launch_bounds__(256) void msconv_kernel(
    const float* __restrict__ x1, const float* __restrict__ x2,
    const float* w3, const float* b3, const float* w5, const float* b5,
    const float* w7, const float* b7, _Float16* __restrict__ xfT) {
    __shared__ float tile[70 * 70];
    __shared__ float effw[49];
    __shared__ float effb;
    int bid = blockIdx.x;
    int c = bid & 255, b = (bid >> 8) & 7, s = bid >> 11;
    const float* x = (s == 0 ? x1 : x2) + ((size_t)(b * CDIM + c)) * NPIX;
    int tid = threadIdx.x;
    for (int i = tid; i < 70 * 70; i += 256) {
        int ty = i / 70, tx = i % 70, gy = ty - 3, gx = tx - 3;
        float vv = 0.f;
        if (gy >= 0 && gy < 64 && gx >= 0 && gx < 64) vv = x[gy * 64 + gx];
        tile[i] = vv;
    }
    if (tid < 49) {
        int dy = tid / 7, dx = tid % 7;
        float w = w7[c * 49 + tid];
        if (dy >= 1 && dy <= 5 && dx >= 1 && dx <= 5) w += w5[c * 25 + (dy - 1) * 5 + (dx - 1)];
        if (dy >= 2 && dy <= 4 && dx >= 2 && dx <= 4) w += w3[c * 9 + (dy - 2) * 3 + (dx - 2)];
        effw[tid] = w * (1.f / 3.f);
    }
    if (tid == 0) effb = (b3[c] + b5[c] + b7[c]) * (1.f / 3.f);
    __syncthreads();
    _Float16* outp = xfT + ((size_t)(s * BATCH + b)) * NPIX * CDIM + c;
#pragma unroll 4
    for (int i = 0; i < 16; ++i) {
        int p = tid + i * 256;
        int y = p >> 6, xx = p & 63;
        float acc = effb;
#pragma unroll
        for (int dy = 0; dy < 7; ++dy)
#pragma unroll
            for (int dx = 0; dx < 7; ++dx)
                acc = fmaf(effw[dy * 7 + dx], tile[(y + dy) * 70 + xx + dx], acc);
        outp[(size_t)p * CDIM] = (_Float16)acc;
    }
}

// BN + exact GELU epilogue store for one accumulator (q/k row-major, v transposed)
__device__ inline void qkv_store(v8f acc, int mbase, int n, const float* __restrict__ al,
                                 const float* __restrict__ be, _Float16* __restrict__ qkb,
                                 _Float16* __restrict__ vb) {
#pragma unroll
    for (int r = 0; r < 8; ++r) {
        int o = mbase + r;
        float val = acc[r] * al[o] + be[o];
        val = 0.5f * val * (1.f + erff(val * 0.70710678f));        // exact GELU
        if (o < 512) {
            qkb[(size_t)o * NPIX + n] = (_Float16)val;
        } else {
            int h = (o - 512) >> 6, dd = (o - 512) & 63;
            vb[(((size_t)h * NPIX) + n) * HDIM + dd] = (_Float16)val;
        }
    }
}

// ---------------- QKV GEMM: Y[768,4096] = Wh(f16) @ xf. One A tile feeds two N tiles.
__global__ __launch_bounds__(256) void qkv_gemm_kernel(
    const _Float16* __restrict__ wh, const _Float16* __restrict__ xfT,
    const float* __restrict__ alpha, const float* __restrict__ beta,
    _Float16* __restrict__ qk, _Float16* __restrict__ vT) {
    int widx = blockIdx.x * 8 + (threadIdx.x >> 5);
    int lane = threadIdx.x & 31;
    int ntp = widx & 127;
    int mt  = (widx >> 7) % 48;
    int sb  = widx / (128 * 48);           // 0..15
    int s = sb >> 3;
    int g = lane >> 4, lm = lane & 15;
    const _Float16* W  = wh + ((size_t)s * O3 + mt * 16 + lm) * CDIM;
    const _Float16* X0 = xfT + ((size_t)sb * NPIX + ntp * 32 + lm) * CDIM;
    const _Float16* X1 = X0 + (size_t)16 * CDIM;
    v8f acc0 = {}, acc1 = {};
#pragma unroll
    for (int k0 = 0; k0 < CDIM; k0 += 32) {
        v16h a = load_a16(W, k0, g);
        acc0 = wmma16(a, load_b16(X0, k0, g), acc0);
        acc1 = wmma16(a, load_b16(X1, k0, g), acc1);
    }
    int n0 = ntp * 32 + lm;
    int mbase = mt * 16 + 8 * g;
    _Float16* qkb = qk + (size_t)sb * 512 * NPIX;
    _Float16* vb  = vT + (size_t)sb * NHEAD * NPIX * HDIM;
    qkv_store(acc0, mbase, n0,      alpha + s * O3, beta + s * O3, qkb, vb);
    qkv_store(acc1, mbase, n0 + 16, alpha + s * O3, beta + s * O3, qkb, vb);
}

// ---------------- Row sum-of-squares -> 1/max(||row||, 1e-12) for all q and k rows
__global__ __launch_bounds__(256) void rownorm_kernel(const _Float16* __restrict__ qk, float* invn) {
    __shared__ float red[256];
    size_t row = blockIdx.x;
    const _Float16* p = qk + row * NPIX;
    float ss = 0.f;
    for (int i = threadIdx.x; i < NPIX; i += 256) { float v = (float)p[i]; ss = fmaf(v, v, ss); }
    red[threadIdx.x] = ss; __syncthreads();
    for (int off = 128; off > 0; off >>= 1) {
        if (threadIdx.x < off) red[threadIdx.x] += red[threadIdx.x + off];
        __syncthreads();
    }
    if (threadIdx.x == 0) invn[row] = 1.f / fmaxf(sqrtf(red[0]), 1e-12f);
}

// ---------------- Attention logits, K split 4 ways: partial S[c,d] = <q_s[c,kc], k_{1-s}[d,kc]>
__global__ __launch_bounds__(256) void attnlog_kernel(
    const _Float16* __restrict__ qk, float* __restrict__ logitp) {
    int widx = blockIdx.x * 8 + (threadIdx.x >> 5);
    int lane = threadIdx.x & 31;
    int ntp = widx & 1, mt = (widx >> 1) & 3, sbh = (widx >> 3) & 63, kc = widx >> 9;
    int s = sbh >> 5, b = (sbh >> 2) & 7, h = sbh & 3;
    int so = 1 - s;
    int g = lane >> 4, lm = lane & 15;
    const _Float16* qrow  = qk + ((size_t)((s * BATCH + b) * 512 + h * HDIM + mt * 16 + lm)) * NPIX;
    const _Float16* krow0 = qk + ((size_t)((so * BATCH + b) * 512 + 256 + h * HDIM + ntp * 32 + lm)) * NPIX;
    const _Float16* krow1 = krow0 + (size_t)16 * NPIX;
    v8f acc0 = {}, acc1 = {};
    int kend = kc * 1024 + 1024;
#pragma unroll 4
    for (int k0 = kc * 1024; k0 < kend; k0 += 32) {
        v16h a = load_a16(qrow, k0, g);
        acc0 = wmma16(a, load_b16(krow0, k0, g), acc0);
        acc1 = wmma16(a, load_b16(krow1, k0, g), acc1);
    }
    int n0 = ntp * 32 + lm;
    float* lrow = logitp + ((size_t)(kc * 64 + sbh)) * HDIM * HDIM;
#pragma unroll
    for (int r = 0; r < 8; ++r) {
        int m = mt * 16 + r + 8 * g;
        lrow[(size_t)m * HDIM + n0]      = acc0[r];
        lrow[(size_t)m * HDIM + n0 + 16] = acc1[r];
    }
}

// ---------------- Sum K-partials, scale by invq*invk*temp, softmax over d (64)
__global__ __launch_bounds__(64) void softmax_kernel(const float* __restrict__ logitp,
                                                     const float* __restrict__ invn,
                                                     _Float16* __restrict__ attn) {
    __shared__ float sm[64];
    int row = blockIdx.x;              // sbh*64 + m
    int sbh = row >> 6, m = row & 63;
    int s = sbh >> 5, b = (sbh >> 2) & 7, h = sbh & 3;
    int tid = threadIdx.x;
    size_t base = (size_t)sbh * HDIM * HDIM + (size_t)m * HDIM + tid;
    float v = logitp[base] + logitp[base + (size_t)64 * HDIM * HDIM]
            + logitp[base + (size_t)128 * HDIM * HDIM] + logitp[base + (size_t)192 * HDIM * HDIM];
    float iq = invn[(size_t)(s * BATCH + b) * 512 + h * HDIM + m];
    float ik = invn[(size_t)((1 - s) * BATCH + b) * 512 + 256 + h * HDIM + tid];
    v *= iq * ik * 0.125f;             // temp = 64^-0.5
    sm[tid] = v; __syncthreads();
    for (int off = 32; off > 0; off >>= 1) {
        if (tid < off) sm[tid] = fmaxf(sm[tid], sm[tid + off]);
        __syncthreads();
    }
    float mx = sm[0]; __syncthreads();
    float e = expf(v - mx);
    sm[tid] = e; __syncthreads();
    for (int off = 32; off > 0; off >>= 1) {
        if (tid < off) sm[tid] += sm[tid + off];
        __syncthreads();
    }
    attn[(size_t)row * 64 + tid] = (_Float16)(e / sm[0]);
}

// ---------------- O = attn(64x64) @ V(64x4096): A frags loaded once, 8 N tiles streamed
__global__ __launch_bounds__(256) void attnout_kernel(
    const _Float16* __restrict__ attn, const _Float16* __restrict__ vT, _Float16* __restrict__ Ot) {
    int widx = blockIdx.x * 8 + (threadIdx.x >> 5);
    int lane = threadIdx.x & 31;
    int ng = widx & 31, mt = (widx >> 5) & 3, h = (widx >> 7) & 3, sb = widx >> 9;
    int g = lane >> 4, lm = lane & 15;
    const _Float16* arow = attn + ((size_t)(sb * NHEAD + h) * HDIM + mt * 16 + lm) * HDIM;
    v16h a0 = load_a16(arow, 0, g);
    v16h a1 = load_a16(arow, 32, g);
    int chb = h * HDIM + mt * 16 + 8 * g;      // 8-aligned -> 16B aligned store
    const _Float16* vbase = vT + ((size_t)(sb * NHEAD + h) * NPIX + lm) * HDIM;
    _Float16* obase = Ot + ((size_t)sb * NPIX + lm) * CDIM + chb;
#pragma unroll
    for (int i = 0; i < 8; ++i) {
        int nt = ng * 8 + i;
        const _Float16* vrow = vbase + (size_t)nt * 16 * HDIM;
        v8f acc = {};
        acc = wmma16(a0, load_b16(vrow, 0, g), acc);
        acc = wmma16(a1, load_b16(vrow, 32, g), acc);
        HPack o;
#pragma unroll
        for (int r = 0; r < 8; ++r) o.h[r] = (_Float16)acc[r];
        *(v8h*)(obase + (size_t)nt * 16 * CDIM) = o.v;
    }
}

// ---------------- SE: global average pool over n
__global__ __launch_bounds__(256) void sepool_kernel(const _Float16* __restrict__ Ot, float* pool) {
    int sb = blockIdx.x, ch = threadIdx.x;
    const _Float16* p = Ot + (size_t)sb * NPIX * CDIM + ch;
    float s = 0.f;
    for (int n = 0; n < NPIX; ++n) s += (float)p[(size_t)n * CDIM];
    pool[sb * CDIM + ch] = s * (1.f / NPIX);
}

// ---------------- SE: two tiny FCs -> sigmoid gate
__global__ __launch_bounds__(256) void semlp_kernel(
    const float* __restrict__ pool, const float* w1, const float* b1,
    const float* w2, const float* b2, float* gate) {
    __shared__ float p[256];
    __shared__ float y[32];
    int sb = blockIdx.x, tid = threadIdx.x;
    p[tid] = pool[sb * 256 + tid]; __syncthreads();
    if (tid < 32) {
        float a = b1[tid];
        for (int c = 0; c < 256; ++c) a = fmaf(w1[tid * 256 + c], p[c], a);
        y[tid] = fmaxf(a, 0.f);
    }
    __syncthreads();
    float a = b2[tid];
    for (int r = 0; r < 32; ++r) a = fmaf(w2[tid * 32 + r], y[r], a);
    gate[sb * 256 + tid] = 1.f / (1.f + expf(-a));
}

// ---------------- Og = Ot * gate (v8h wide)
__global__ __launch_bounds__(256) void gateapply_kernel(
    const _Float16* __restrict__ Ot, const float* __restrict__ gate, _Float16* __restrict__ Og) {
    int idx = blockIdx.x * 256 + threadIdx.x;
    int chg = idx & 31, n = (idx >> 5) & 4095, sb = idx >> 17;
    size_t off = ((size_t)sb * NPIX + n) * CDIM + chg * 8;
    const float* gt = gate + sb * CDIM + chg * 8;
    HPack v; v.v = *(const v8h*)(Ot + off);
#pragma unroll
    for (int j = 0; j < 8; ++j) v.h[j] = (_Float16)((float)v.h[j] * gt[j]);
    *(v8h*)(Og + off) = v.v;
}

// ---------------- Output projection: out = Wp(f16) @ Og + bias + residual (fp32 out)
__global__ __launch_bounds__(256) void proj_kernel(
    const _Float16* __restrict__ Og, const _Float16* __restrict__ wph,
    const float* bb1, const float* bb2,
    const float* __restrict__ x1, const float* __restrict__ x2, float* __restrict__ out) {
    int widx = blockIdx.x * 8 + (threadIdx.x >> 5);
    int lane = threadIdx.x & 31;
    int ntp = widx & 127, mt = (widx >> 7) & 15, sb = widx >> 11;
    int b = sb & 7, s = sb >> 3;
    int g = lane >> 4, lm = lane & 15;
    const _Float16* W  = wph + ((size_t)s * CDIM + mt * 16 + lm) * CDIM;
    const _Float16* B0 = Og + ((size_t)sb * NPIX + ntp * 32 + lm) * CDIM;
    const _Float16* B1 = B0 + (size_t)16 * CDIM;
    v8f acc0 = {}, acc1 = {};
#pragma unroll
    for (int k0 = 0; k0 < CDIM; k0 += 32) {
        v16h a = load_a16(W, k0, g);
        acc0 = wmma16(a, load_b16(B0, k0, g), acc0);
        acc1 = wmma16(a, load_b16(B1, k0, g), acc1);
    }
    int n0 = ntp * 32 + lm;
    const float* bias = s ? bb2 : bb1;
    const float* xin = (s ? x2 : x1) + ((size_t)b * CDIM) * NPIX;
    float* op = out + ((size_t)s * BATCH + b) * CDIM * NPIX;
#pragma unroll
    for (int r = 0; r < 8; ++r) {
        int o = mt * 16 + r + 8 * g;
        op[(size_t)o * NPIX + n0]      = acc0[r] + bias[o] + xin[(size_t)o * NPIX + n0];
        op[(size_t)o * NPIX + n0 + 16] = acc1[r] + bias[o] + xin[(size_t)o * NPIX + n0 + 16];
    }
}

extern "C" void kernel_launch(void* const* d_in, const int* in_sizes, int n_in,
                              void* d_out, int out_size, void* d_ws, size_t ws_size,
                              hipStream_t stream) {
    const float* x1    = (const float*)d_in[0];
    const float* x2    = (const float*)d_in[1];
    const float* ms_w3 = (const float*)d_in[2];
    const float* ms_b3 = (const float*)d_in[3];
    const float* ms_w5 = (const float*)d_in[4];
    const float* ms_b5 = (const float*)d_in[5];
    const float* ms_w7 = (const float*)d_in[6];
    const float* ms_b7 = (const float*)d_in[7];
    const float* qkv1_w = (const float*)d_in[8];
    const float* qkv1_b = (const float*)d_in[9];
    const float* bn1_g  = (const float*)d_in[10];
    const float* bn1_b  = (const float*)d_in[11];
    const float* bn1_m  = (const float*)d_in[12];
    const float* bn1_v  = (const float*)d_in[13];
    const float* qkv2_w = (const float*)d_in[14];
    const float* qkv2_b = (const float*)d_in[15];
    const float* bn2_g  = (const float*)d_in[16];
    const float* bn2_b  = (const float*)d_in[17];
    const float* bn2_m  = (const float*)d_in[18];
    const float* bn2_v  = (const float*)d_in[19];
    const float* ca_w1  = (const float*)d_in[20];
    const float* ca_b1  = (const float*)d_in[21];
    const float* ca_w2  = (const float*)d_in[22];
    const float* ca_b2  = (const float*)d_in[23];
    const float* po1_w  = (const float*)d_in[24];
    const float* po1_b  = (const float*)d_in[25];
    const float* po2_w  = (const float*)d_in[26];
    const float* po2_b  = (const float*)d_in[27];

    // workspace carve-up (all offsets 256B aligned)
    char* ws = (char*)d_ws;
    size_t off = 0;
    auto take = [&](size_t bytes) { char* p = ws + off; off += (bytes + 255) & ~(size_t)255; return p; };
    _Float16* xfT    = (_Float16*)take((size_t)NSTREAM * BATCH * NPIX * CDIM * 2);          // 33.5 MB
    _Float16* qk     = (_Float16*)take((size_t)NSTREAM * BATCH * 512 * NPIX * 2);           // 67 MB
    _Float16* vT     = (_Float16*)take((size_t)NSTREAM * BATCH * NHEAD * NPIX * HDIM * 2);  // 33.5 MB
    float*    invn   = (float*)take((size_t)NSTREAM * BATCH * 512 * 4);
    float*    logitp = (float*)take((size_t)4 * 64 * HDIM * HDIM * 4);                      // 4 MB
    _Float16* attn   = (_Float16*)take((size_t)NSTREAM * BATCH * NHEAD * HDIM * HDIM * 2);
    _Float16* Ot     = (_Float16*)take((size_t)NSTREAM * BATCH * NPIX * CDIM * 2);          // 33.5 MB
    _Float16* Og     = (_Float16*)take((size_t)NSTREAM * BATCH * NPIX * CDIM * 2);          // 33.5 MB
    float*    pool   = (float*)take((size_t)NSTREAM * BATCH * CDIM * 4);
    float*    gate   = (float*)take((size_t)NSTREAM * BATCH * CDIM * 4);
    float*    alpha  = (float*)take((size_t)NSTREAM * O3 * 4);
    float*    beta   = (float*)take((size_t)NSTREAM * O3 * 4);
    _Float16* whq    = (_Float16*)take((size_t)NSTREAM * O3 * CDIM * 2);                    // 786 KB
    _Float16* whp    = (_Float16*)take((size_t)NSTREAM * CDIM * CDIM * 2);                  // 262 KB
    (void)ws_size; (void)in_sizes; (void)n_in; (void)out_size;

    bnprep_kernel<<<2, 768, 0, stream>>>(qkv1_b, bn1_g, bn1_b, bn1_m, bn1_v,
                                         qkv2_b, bn2_g, bn2_b, bn2_m, bn2_v, alpha, beta);
    cvt16_kernel<<<(O3 * CDIM) / 1024, 256, 0, stream>>>(qkv1_w, whq, O3 * CDIM);
    cvt16_kernel<<<(O3 * CDIM) / 1024, 256, 0, stream>>>(qkv2_w, whq + (size_t)O3 * CDIM, O3 * CDIM);
    cvt16_kernel<<<(CDIM * CDIM) / 1024, 256, 0, stream>>>(po1_w, whp, CDIM * CDIM);
    cvt16_kernel<<<(CDIM * CDIM) / 1024, 256, 0, stream>>>(po2_w, whp + (size_t)CDIM * CDIM, CDIM * CDIM);
    msconv_kernel<<<NSTREAM * BATCH * CDIM, 256, 0, stream>>>(x1, x2, ms_w3, ms_b3, ms_w5, ms_b5,
                                                              ms_w7, ms_b7, xfT);
    qkv_gemm_kernel<<<(16 * 48 * 128) / 8, 256, 0, stream>>>(whq, xfT, alpha, beta, qk, vT);
    rownorm_kernel<<<NSTREAM * BATCH * 512, 256, 0, stream>>>(qk, invn);
    attnlog_kernel<<<(4 * 64 * 4 * 2) / 8, 256, 0, stream>>>(qk, logitp);
    softmax_kernel<<<64 * HDIM, 64, 0, stream>>>(logitp, invn, attn);
    attnout_kernel<<<(16 * 4 * 4 * 32) / 8, 256, 0, stream>>>(attn, vT, Ot);
    sepool_kernel<<<NSTREAM * BATCH, 256, 0, stream>>>(Ot, pool);
    semlp_kernel<<<NSTREAM * BATCH, 256, 0, stream>>>(pool, ca_w1, ca_b1, ca_w2, ca_b2, gate);
    gateapply_kernel<<<(NSTREAM * BATCH * NPIX * 32) / 256, 256, 0, stream>>>(Ot, gate, Og);
    proj_kernel<<<(16 * 16 * 128) / 8, 256, 0, stream>>>(Og, whp, po1_b, po2_b, x1, x2,
                                                         (float*)d_out);
}
